// DiffusionBlock_43963285242226
// MI455X (gfx1250) — matrix operand order repaired and checked by reading, not compiled, not run on hardware
//
#include <hip/hip_runtime.h>
#include <hip/hip_bf16.h>

// ---------------------------------------------------------------------------
// DiffusionBlock for MI455X (gfx1250): all GEMM-like work on v_wmma_f32_16x16x32_bf16.
// B=8, N=1024, D=1024, H=16 (dh=64), CH=8 (dh_c=128).
// ---------------------------------------------------------------------------

typedef __bf16 bf16;
typedef __attribute__((ext_vector_type(16))) __bf16 v16bf;
typedef __attribute__((ext_vector_type(8)))  float   v8f;

union FragBF { v16bf v; uint4 q[2]; };

__device__ __forceinline__ v8f zero8() {
    v8f z = {0.f,0.f,0.f,0.f,0.f,0.f,0.f,0.f};
    return z;
}

__device__ __forceinline__ v8f wmma_bf16(v16bf a, v16bf b, v8f c) {
    return __builtin_amdgcn_wmma_f32_16x16x32_bf16(
        /*neg_a=*/false, a, /*neg_b=*/false, b,
        /*c_mod=*/(short)0, c, /*reuse_a=*/false, /*reuse_b=*/false);
}

// A fragment: 16x32 bf16 tile, row-major source, row stride lda (elements).
__device__ __forceinline__ v16bf load_a_frag(const bf16* A, int lda, int lane, int kofs) {
    int r  = lane & 15;
    int kh = (lane >> 4) * 8;
    const bf16* p = A + (size_t)r * lda + kofs + kh;
    FragBF f;
    f.q[0] = *(const uint4*)(p);
    f.q[1] = *(const uint4*)(p + 16);
    return f.v;
}

// B fragment: 32(K)x16(N) bf16. Bcm is "column-major by N": Bcm[n][k], row stride ldb.
__device__ __forceinline__ v16bf load_b_frag(const bf16* Bcm, int ldb, int lane, int kofs) {
    int n  = lane & 15;
    int kh = (lane >> 4) * 8;
    const bf16* p = Bcm + (size_t)n * ldb + kofs + kh;
    FragBF f;
    f.q[0] = *(const uint4*)(p);
    f.q[1] = *(const uint4*)(p + 16);
    return f.v;
}

// A fragment from f32 LDS (softmax probabilities) via b128 LDS reads + cvt.
__device__ __forceinline__ v16bf load_a_frag_lds_f32(const float* S, int lds, int lane, int kofs) {
    int r  = lane & 15;
    int kh = (lane >> 4) * 8;
    const float* p = S + r * lds + kofs + kh;
    float4 a0 = *(const float4*)(p);
    float4 a1 = *(const float4*)(p + 4);
    float4 b0 = *(const float4*)(p + 16);
    float4 b1 = *(const float4*)(p + 20);
    v16bf f;
    f[0]  = (bf16)a0.x; f[1]  = (bf16)a0.y; f[2]  = (bf16)a0.z; f[3]  = (bf16)a0.w;
    f[4]  = (bf16)a1.x; f[5]  = (bf16)a1.y; f[6]  = (bf16)a1.z; f[7]  = (bf16)a1.w;
    f[8]  = (bf16)b0.x; f[9]  = (bf16)b0.y; f[10] = (bf16)b0.z; f[11] = (bf16)b0.w;
    f[12] = (bf16)b1.x; f[13] = (bf16)b1.y; f[14] = (bf16)b1.z; f[15] = (bf16)b1.w;
    return f;
}

__device__ __forceinline__ unsigned pack2(float a, float b) {
    union { bf16 h[2]; unsigned u; } p;
    p.h[0] = (bf16)a; p.h[1] = (bf16)b;
    return p.u;
}

// ---------------------------------------------------------------------------
// Weight convert + transpose, LDS-tiled (64x64): WT[n*D+k] = (bf16)W[k*D+n]
// ---------------------------------------------------------------------------
__global__ void k_cvt_wT(const float* __restrict__ W, bf16* __restrict__ WT) {
    __shared__ float tile[64][68];
    int kb = blockIdx.x * 64, nb = blockIdx.y * 64;
    int tid = threadIdx.x;
    int c4 = (tid & 15) * 4;
    int rl = tid >> 4;
#pragma unroll
    for (int p = 0; p < 4; ++p) {
        int k = rl + p * 16;
        float4 d = *(const float4*)(W + (size_t)(kb + k) * 1024 + nb + c4);
        tile[k][c4 + 0] = d.x; tile[k][c4 + 1] = d.y;
        tile[k][c4 + 2] = d.z; tile[k][c4 + 3] = d.w;
    }
    __syncthreads();
#pragma unroll
    for (int p = 0; p < 4; ++p) {
        int n = rl + p * 16;
        union { bf16 h[4]; uint2 u; } pk;
#pragma unroll
        for (int i = 0; i < 4; ++i) pk.h[i] = (bf16)tile[c4 + i][n];
        *(uint2*)(WT + (size_t)(nb + n) * 1024 + kb + c4) = pk.u;
    }
}

// ---------------------------------------------------------------------------
// bf16 transpose (B,N,D)->(B,D,N), LDS-tiled 64x64. grid (16,16,8), 256 thr.
// ---------------------------------------------------------------------------
__global__ void k_transpose(const bf16* __restrict__ in, bf16* __restrict__ out) {
    __shared__ bf16 tile[64][72];
    int cb = blockIdx.x * 64, nb = blockIdx.y * 64, b = blockIdx.z;
    int tid = threadIdx.x;
    int c4 = (tid & 15) * 4;
    int rl = tid >> 4;
    const bf16* src = in + ((size_t)b * 1024 + nb) * 1024 + cb;
#pragma unroll
    for (int p = 0; p < 4; ++p) {
        int n = rl + p * 16;
        *(uint2*)&tile[n][c4] = *(const uint2*)(src + (size_t)n * 1024 + c4);
    }
    __syncthreads();
    bf16* dst = out + ((size_t)b * 1024 + cb) * 1024 + nb;
#pragma unroll
    for (int p = 0; p < 4; ++p) {
        int c = rl + p * 16;
        union { bf16 h[4]; uint2 u; } pk;
#pragma unroll
        for (int i = 0; i < 4; ++i) pk.h[i] = tile[c4 + i][c];
        *(uint2*)(dst + (size_t)c * 1024 + c4) = pk.u;
    }
}

// ---------------------------------------------------------------------------
// LayerNorm: one block (256 thr) per row of 1024, output bf16.
// ---------------------------------------------------------------------------
__global__ void k_ln(const float* __restrict__ x, const float* __restrict__ w,
                     const float* __restrict__ b, bf16* __restrict__ y) {
    int row = blockIdx.x, tid = threadIdx.x;
    const float* xr = x + (size_t)row * 1024;
    __shared__ float red[256];
    float4 d0 = *(const float4*)(xr + tid * 4);
    float s = d0.x + d0.y + d0.z + d0.w;
    red[tid] = s; __syncthreads();
    for (int o = 128; o; o >>= 1) { if (tid < o) red[tid] += red[tid + o]; __syncthreads(); }
    float mu = red[0] * (1.f / 1024.f);
    __syncthreads();
    float vx = (d0.x - mu) * (d0.x - mu) + (d0.y - mu) * (d0.y - mu)
             + (d0.z - mu) * (d0.z - mu) + (d0.w - mu) * (d0.w - mu);
    red[tid] = vx; __syncthreads();
    for (int o = 128; o; o >>= 1) { if (tid < o) red[tid] += red[tid + o]; __syncthreads(); }
    float rstd = rsqrtf(red[0] * (1.f / 1024.f) + 1e-5f);
    float4 wv = *(const float4*)(w + tid * 4);
    float4 bv = *(const float4*)(b + tid * 4);
    union { bf16 h[4]; uint2 u; } pk;
    pk.h[0] = (bf16)((d0.x - mu) * rstd * wv.x + bv.x);
    pk.h[1] = (bf16)((d0.y - mu) * rstd * wv.y + bv.y);
    pk.h[2] = (bf16)((d0.z - mu) * rstd * wv.z + bv.z);
    pk.h[3] = (bf16)((d0.w - mu) * rstd * wv.w + bv.w);
    *(uint2*)(y + (size_t)row * 1024 + tid * 4) = pk.u;
}

// ---------------------------------------------------------------------------
// Sum of squares over contiguous rows of bf16 (len % 4 == 0).
// ---------------------------------------------------------------------------
__global__ void k_rowsumsq(const bf16* __restrict__ in, float* __restrict__ out,
                           int len, int nrows) {
    int row = blockIdx.x * blockDim.x + threadIdx.x;
    if (row >= nrows) return;
    const uint2* p = (const uint2*)(in + (size_t)row * len);
    float s = 0.f;
    for (int i = 0; i < (len >> 2); ++i) {
        union { uint2 u; bf16 h[4]; } d; d.u = p[i];
#pragma unroll
        for (int j = 0; j < 4; ++j) { float v = (float)d.h[j]; s += v * v; }
    }
    out[row] = s;
}

// ---------------------------------------------------------------------------
// GEMM: C(8192 x 1024) = A(bf16 rowmajor) @ B, Bcm[n][k] = transposed weights.
// 256 threads = 8 waves (4 row x 2 col); wave tile 16x64 (4 accs);
// block tile 64x128; grid (128, 8). Ping-pong K loop, unroll 2 (no reg copies).
// ---------------------------------------------------------------------------
template<bool RESID>
__device__ __forceinline__ void gemm_body(const bf16* __restrict__ A,
                                          const bf16* __restrict__ Bcm,
                                          bf16* __restrict__ C,
                                          const float* __restrict__ resid,
                                          const float* __restrict__ gamma,
                                          float* __restrict__ outf) {
    const int K = 1024;
    int tid = threadIdx.x, wave = tid >> 5, lane = tid & 31;
    int wrow = wave >> 1, wcol = wave & 1;
    size_t rbase = (size_t)blockIdx.x * 64 + wrow * 16;
    size_t cbase = (size_t)blockIdx.y * 128 + wcol * 64;
    const bf16* arow = A + rbase * K;

    v8f acc[4];
#pragma unroll
    for (int t = 0; t < 4; ++t) acc[t] = zero8();

    v16bf aX = load_a_frag(arow, K, lane, 0);
    v16bf bX[4];
#pragma unroll
    for (int t = 0; t < 4; ++t) bX[t] = load_b_frag(Bcm + (cbase + t * 16) * K, K, lane, 0);

    for (int kofs = 0; kofs < K; kofs += 64) {
        // prefetch set Y (kofs+32) while WMMAs consume set X
        v16bf aY = load_a_frag(arow, K, lane, kofs + 32);
        v16bf bY[4];
#pragma unroll
        for (int t = 0; t < 4; ++t)
            bY[t] = load_b_frag(Bcm + (cbase + t * 16) * K, K, lane, kofs + 32);
#pragma unroll
        for (int t = 0; t < 4; ++t) acc[t] = wmma_bf16(aX, bX[t], acc[t]);
        int k2 = kofs + 64;
        if (k2 < K) {                       // uniform; prefetch set X (kofs+64)
            aX = load_a_frag(arow, K, lane, k2);
#pragma unroll
            for (int t = 0; t < 4; ++t)
                bX[t] = load_b_frag(Bcm + (cbase + t * 16) * K, K, lane, k2);
        }
#pragma unroll
        for (int t = 0; t < 4; ++t) acc[t] = wmma_bf16(aY, bY[t], acc[t]);
    }

    int n = lane & 15, hi = lane >> 4;
#pragma unroll
    for (int t = 0; t < 4; ++t) {
        size_t col = cbase + t * 16 + n;
        if (RESID) {
            float g = gamma[col];
#pragma unroll
            for (int r = 0; r < 8; ++r) {
                int m = r + 8 * hi;
                size_t i0 = (rbase + m) * 1024 + col;
                outf[i0] = resid[i0] + acc[t][r] * g;
            }
        } else {
#pragma unroll
            for (int r = 0; r < 8; ++r) {
                int m = r + 8 * hi;
                C[(rbase + m) * 1024 + col] = (bf16)acc[t][r];
            }
        }
    }
}

__global__ void __launch_bounds__(256, 1)
k_gemm_bf16(const bf16* __restrict__ A, const bf16* __restrict__ Bcm,
            bf16* __restrict__ C) {
    gemm_body<false>(A, Bcm, C, nullptr, nullptr, nullptr);
}

__global__ void __launch_bounds__(256, 1)
k_gemm_resid(const bf16* __restrict__ A, const bf16* __restrict__ Bcm,
             const float* __restrict__ resid, const float* __restrict__ gamma,
             float* __restrict__ out) {
    gemm_body<true>(A, Bcm, nullptr, resid, gamma, out);
}

// ---------------------------------------------------------------------------
// Token diffusion attention. One block (4 waves) per (b, h, 16-row tile).
// qk: (B,N,D) bf16; vt: head-transposed V, row (b*1024 + h*64 + d) of length N.
// q2: [(b*N + j)*H + h]. (-q2_i is row-constant -> cancels in softmax.)
// ---------------------------------------------------------------------------
__global__ void __launch_bounds__(128, 1)
k_token_attn(const bf16* __restrict__ qk, const bf16* __restrict__ vt,
             const float* __restrict__ q2, const float* __restrict__ tau,
             bf16* __restrict__ ao) {
    const int N = 1024, D = 1024, H = 16;
    int bid = blockIdx.x;
    int it = bid & 63, h = (bid >> 6) & 15, b = bid >> 10;
    int tid = threadIdx.x, wave = tid >> 5, lane = tid & 31;
    __shared__ float sS[16 * 1024];                       // 64 KB scores

    const bf16* qbase = qk + ((size_t)b * N) * D + h * 64;
    const bf16* arow = qbase + (size_t)(it * 16) * D;
    v16bf fa0 = load_a_frag(arow, D, lane, 0);
    v16bf fa1 = load_a_frag(arow, D, lane, 32);
    float sc = tau[h] * 0.125f;                           // tau/sqrt(64)
    int n = lane & 15, hi = lane >> 4;

    int jt0 = wave * 16;
    v16bf x0 = load_b_frag(qbase + (size_t)(jt0 * 16) * D, D, lane, 0);
    v16bf x1 = load_b_frag(qbase + (size_t)(jt0 * 16) * D, D, lane, 32);
    for (int jt = jt0; jt < jt0 + 16; jt += 2) {
        const bf16* rowY = qbase + (size_t)((jt + 1) * 16) * D;
        v16bf y0 = load_b_frag(rowY, D, lane, 0);
        v16bf y1 = load_b_frag(rowY, D, lane, 32);
        {
            v8f acc = zero8();
            acc = wmma_bf16(fa0, x0, acc);
            acc = wmma_bf16(fa1, x1, acc);
            int j = jt * 16 + n;
            float q2j = q2[((size_t)b * N + j) * H + h];
#pragma unroll
            for (int r = 0; r < 8; ++r)
                sS[(r + 8 * hi) * 1024 + j] = sc * (2.0f * acc[r] - q2j);
        }
        if (jt + 2 < jt0 + 16) {            // uniform
            const bf16* rowX = qbase + (size_t)((jt + 2) * 16) * D;
            x0 = load_b_frag(rowX, D, lane, 0);
            x1 = load_b_frag(rowX, D, lane, 32);
        }
        {
            v8f acc = zero8();
            acc = wmma_bf16(fa0, y0, acc);
            acc = wmma_bf16(fa1, y1, acc);
            int j = (jt + 1) * 16 + n;
            float q2j = q2[((size_t)b * N + j) * H + h];
#pragma unroll
            for (int r = 0; r < 8; ++r)
                sS[(r + 8 * hi) * 1024 + j] = sc * (2.0f * acc[r] - q2j);
        }
    }
    __syncthreads();

    // softmax: 8 consecutive lanes per row (shuffle reduction, width 8)
    int rr = tid >> 3, tsub = tid & 7;
    float mx = -3.0e38f;
    for (int k = 0; k < 128; ++k) mx = fmaxf(mx, sS[rr * 1024 + tsub + 8 * k]);
    for (int o = 4; o; o >>= 1) mx = fmaxf(mx, __shfl_xor(mx, o, 8));
    float sum = 0.f;
    for (int k = 0; k < 128; ++k) {
        int j = tsub + 8 * k;
        float e = __expf(sS[rr * 1024 + j] - mx);
        sS[rr * 1024 + j] = e;
        sum += e;
    }
    for (int o = 4; o; o >>= 1) sum += __shfl_xor(sum, o, 8);
    float inv = 1.0f / sum;
    for (int k = 0; k < 128; ++k) sS[rr * 1024 + tsub + 8 * k] *= inv;
    __syncthreads();

    // out(16 x 64) = P(16x1024) @ V ; wave handles columns d = wave*16..+15
    const bf16* vbase = vt + ((size_t)b * D + h * 64 + wave * 16) * 1024;
    v8f acc = zero8();
    v16bf bX = load_b_frag(vbase, 1024, lane, 0);
    for (int kofs = 0; kofs < 1024; kofs += 64) {
        v16bf bY = load_b_frag(vbase, 1024, lane, kofs + 32);
        v16bf aX = load_a_frag_lds_f32(sS, 1024, lane, kofs);
        acc = wmma_bf16(aX, bX, acc);
        if (kofs + 64 < 1024) bX = load_b_frag(vbase, 1024, lane, kofs + 64);
        v16bf aY = load_a_frag_lds_f32(sS, 1024, lane, kofs + 32);
        acc = wmma_bf16(aY, bY, acc);
    }
    size_t col = h * 64 + wave * 16 + n;
#pragma unroll
    for (int r = 0; r < 8; ++r) {
        int m = r + 8 * hi;
        ao[((size_t)b * N + it * 16 + m) * D + col] = (bf16)acc[r];
    }
}

// ---------------------------------------------------------------------------
// Channel diffusion attention. One block per (b, ch, 16-channel tile).
// qkct: (B,D,N) bf16 (channels are rows); vc: (B,N,D) bf16 (already the
// column-major-by-n operand for P@V); q2c: [b*D + c].
// ---------------------------------------------------------------------------
__global__ void __launch_bounds__(128, 1)
k_chan_attn(const bf16* __restrict__ qkct, const bf16* __restrict__ vc,
            const float* __restrict__ q2c, const float* __restrict__ tau,
            bf16* __restrict__ ao) {
    const int N = 1024, D = 1024;
    int bid = blockIdx.x;
    int it = bid & 7, ch = (bid >> 3) & 7, b = bid >> 6;
    int tid = threadIdx.x, wave = tid >> 5, lane = tid & 31;
    __shared__ float sS[16 * 128];                        // 8 KB scores

    const bf16* qbase = qkct + ((size_t)b * D + ch * 128) * 1024;
    const bf16* arow = qbase + (size_t)(it * 16) * 1024;
    float sc = tau[ch] * (1.0f / 32.0f);                  // tau/sqrt(1024)
    int n = lane & 15, hi = lane >> 4;

    for (int jt = wave * 2; jt < wave * 2 + 2; ++jt) {
        const bf16* brow = qbase + (size_t)(jt * 16) * 1024;
        v8f acc = zero8();
        v16bf aX = load_a_frag(arow, 1024, lane, 0);
        v16bf bX = load_b_frag(brow, 1024, lane, 0);
        for (int kofs = 0; kofs < 1024; kofs += 64) {
            v16bf aY = load_a_frag(arow, 1024, lane, kofs + 32);
            v16bf bY = load_b_frag(brow, 1024, lane, kofs + 32);
            acc = wmma_bf16(aX, bX, acc);
            if (kofs + 64 < 1024) {
                aX = load_a_frag(arow, 1024, lane, kofs + 64);
                bX = load_b_frag(brow, 1024, lane, kofs + 64);
            }
            acc = wmma_bf16(aY, bY, acc);
        }
        int j = jt * 16 + n;
        float q2j = q2c[(size_t)b * D + ch * 128 + j];
#pragma unroll
        for (int r = 0; r < 8; ++r)
            sS[(r + 8 * hi) * 128 + j] = sc * (2.0f * acc[r] - q2j);
    }
    __syncthreads();

    int rr = tid >> 3, tsub = tid & 7;
    float mx = -3.0e38f;
    for (int k = 0; k < 16; ++k) mx = fmaxf(mx, sS[rr * 128 + tsub + 8 * k]);
    for (int o = 4; o; o >>= 1) mx = fmaxf(mx, __shfl_xor(mx, o, 8));
    float sum = 0.f;
    for (int k = 0; k < 16; ++k) {
        int j = tsub + 8 * k;
        float e = __expf(sS[rr * 128 + j] - mx);
        sS[rr * 128 + j] = e;
        sum += e;
    }
    for (int o = 4; o; o >>= 1) sum += __shfl_xor(sum, o, 8);
    float inv = 1.0f / sum;
    for (int k = 0; k < 16; ++k) sS[rr * 128 + tsub + 8 * k] *= inv;
    __syncthreads();

    // out(16 channels x 1024 tokens) = P(16x128) @ Vt ; K=128, hoist A frags.
    v16bf fa[4];
#pragma unroll
    for (int kk = 0; kk < 4; ++kk) fa[kk] = load_a_frag_lds_f32(sS, 128, lane, kk * 32);
    const bf16* vbase = vc + (size_t)b * N * D + ch * 128;   // Bcm row n: vbase + n*D
    for (int nt = wave * 16; nt < wave * 16 + 16; ++nt) {
        const bf16* brow = vbase + (size_t)(nt * 16) * D;
        v16bf fb[4];
#pragma unroll
        for (int kk = 0; kk < 4; ++kk) fb[kk] = load_b_frag(brow, D, lane, kk * 32);
        v8f acc = zero8();
#pragma unroll
        for (int kk = 0; kk < 4; ++kk) acc = wmma_bf16(fa[kk], fb[kk], acc);
        size_t nglob = nt * 16 + n;
        uint4 pkt;
        pkt.x = pack2(acc[0], acc[1]);
        pkt.y = pack2(acc[2], acc[3]);
        pkt.z = pack2(acc[4], acc[5]);
        pkt.w = pack2(acc[6], acc[7]);
        *(uint4*)&ao[((size_t)b * N + nglob) * D + ch * 128 + it * 16 + 8 * hi] = pkt;
    }
}

// ---------------------------------------------------------------------------
// Host-side launcher
// ---------------------------------------------------------------------------
extern "C" void kernel_launch(void* const* d_in, const int* in_sizes, int n_in,
                              void* d_out, int out_size, void* d_ws, size_t ws_size,
                              hipStream_t stream) {
    const float* x     = (const float*)d_in[0];
    const float* tau_t = (const float*)d_in[4];
    const float* tau_c = (const float*)d_in[8];
    const float* ln1w  = (const float*)d_in[9];
    const float* ln1b  = (const float*)d_in[10];
    const float* ln2w  = (const float*)d_in[11];
    const float* ln2b  = (const float*)d_in[12];
    const float* g1    = (const float*)d_in[13];
    const float* g2    = (const float*)d_in[14];
    float* out = (float*)d_out;

    const size_t DD = 1024ull * 1024ull;        // 1M
    const size_t MD = 8ull * 1024ull * 1024ull; // 8M (B*N*D)

    bf16* wsb = (bf16*)d_ws;
    bf16* wT[6];
    for (int i = 0; i < 6; ++i) wT[i] = wsb + i * DD;  // 12 MB transposed bf16 weights
    bf16* xn = wsb + 6 * DD;                            // 16 MB LN output
    bf16* qk = xn + MD;                                 // 16 MB
    bf16* v  = qk + MD;                                 // 16 MB
    bf16* tr = v + MD;                                  // 16 MB transpose buffer
    bf16* ao = tr + MD;                                 // 16 MB attention output
    float* q2 = (float*)(ao + MD);                      // 512 KB

    const float* Ws[6] = { (const float*)d_in[1], (const float*)d_in[2], (const float*)d_in[3],
                           (const float*)d_in[5], (const float*)d_in[6], (const float*)d_in[7] };
    for (int i = 0; i < 6; ++i)
        k_cvt_wT<<<dim3(16, 16), 256, 0, stream>>>(Ws[i], wT[i]);

    dim3 gg(128, 8);          // (8192/64, 1024/128), 256 threads = 8 waves
    dim3 gt(16, 16, 8);       // transpose tiles

    // ---- token diffusion sublayer ----
    k_ln<<<8192, 256, 0, stream>>>(x, ln1w, ln1b, xn);
    k_gemm_bf16<<<gg, 256, 0, stream>>>(xn, wT[0], qk);
    k_gemm_bf16<<<gg, 256, 0, stream>>>(xn, wT[1], v);
    k_rowsumsq<<<131072 / 256, 256, 0, stream>>>(qk, q2, 64, 131072);   // per (b,n,h)
    k_transpose<<<gt, 256, 0, stream>>>(v, tr);                         // head-transposed V
    k_token_attn<<<8192, 128, 0, stream>>>(qk, tr, q2, tau_t, ao);
    k_gemm_resid<<<gg, 256, 0, stream>>>(ao, wT[2], x, g1, out);        // y = x + (ao@Wo)*g1

    // ---- channel diffusion sublayer ----
    k_ln<<<8192, 256, 0, stream>>>(out, ln2w, ln2b, xn);
    k_gemm_bf16<<<gg, 256, 0, stream>>>(xn, wT[3], qk);
    k_gemm_bf16<<<gg, 256, 0, stream>>>(xn, wT[4], v);
    k_transpose<<<gt, 256, 0, stream>>>(qk, tr);                        // (B,D,N) channels
    k_rowsumsq<<<8192 / 256, 256, 0, stream>>>(tr, q2, 1024, 8192);     // per (b,c)
    k_chan_attn<<<512, 128, 0, stream>>>(tr, v, q2, tau_c, ao);
    k_gemm_resid<<<gg, 256, 0, stream>>>(ao, wT[5], out, g2, out);      // in-place residual
}
